// ConditionalConv11_3075196584424
// MI455X (gfx1250) — matrix operand order repaired and checked by reading, not compiled, not run on hardware
//
#include <hip/hip_runtime.h>

typedef __attribute__((ext_vector_type(16))) __bf16 v16bf;
typedef __attribute__((ext_vector_type(8)))  __bf16 v8bf;
typedef __attribute__((ext_vector_type(4)))  __bf16 v4bf;
typedef __attribute__((ext_vector_type(8)))  float  v8f;

#define IN_C     512
#define FILTERS  512
#define PIX      4096   // 64*64 pixels per sample
#define NUMCLS   120
#define MT       128    // M tile per workgroup
#define NT       128    // N tile per workgroup
#define KT       32     // K slice per LDS stage
#define LDK      40     // padded LDS row stride (bf16 elems) = 80B, 16B aligned
#define NKSTEPS  (IN_C / KT)

__device__ __forceinline__ void split_bf16(float f, __bf16& hi, __bf16& lo) {
    hi = (__bf16)f;                 // RNE: top ~8 mantissa bits
    lo = (__bf16)(f - (float)hi);   // residual: next ~8 mantissa bits
}

union AFrag { v16bf v; v8bf h[2]; };

__global__ __launch_bounds__(256) void cond_conv11_wmma_bf16x3(
    const float* __restrict__ x,      // [B, PIX, IN_C]
    const int*   __restrict__ cls,    // [B]
    const float* __restrict__ ktab,   // [NUMCLS, IN_C, FILTERS]
    const float* __restrict__ bias,   // [NUMCLS, FILTERS]
    float*       __restrict__ out)    // [B, PIX, FILTERS]
{
    __shared__ __bf16 As_hi[MT * LDK];   // x tile, hi part, [m][k]
    __shared__ __bf16 As_lo[MT * LDK];   // x tile, lo part
    __shared__ __bf16 Wt_hi[NT * LDK];   // weight tile TRANSPOSED, [n][k]
    __shared__ __bf16 Wt_lo[NT * LDK];

    const int tid  = threadIdx.x;
    const int lane = tid & 31;
    const int wv   = tid >> 5;          // wave id 0..7
    const int b    = blockIdx.z;
    const int bm   = blockIdx.y * MT;
    const int bn   = blockIdx.x * NT;

    int c = cls[b];
    c = (c < 0) ? 0 : (c >= NUMCLS ? NUMCLS - 1 : c);

    const float* xb = x    + ((size_t)b * PIX + bm) * IN_C;          // [128 rows][512]
    const float* wb = ktab + (size_t)c * IN_C * FILTERS + bn;        // [512 rows][128 cols]

    // wave sub-tile: 4 waves along M (32 rows each), 2 along N (64 cols each)
    const int wm  = (wv & 3) * 32;
    const int wn  = (wv >> 2) * 64;
    const int h   = lane >> 4;          // lane half (ISA K/M-half select)
    const int l16 = lane & 15;

    v8f acc[2][4] = {};

    // ---- global-load helpers (register prefetch buffers) ----
    float4 xr[4], wr[4];

    auto load_x = [&](int kt) {
        const float* base = xb + kt * KT;
        #pragma unroll
        for (int i = 0; i < 4; ++i) {
            int idx = tid + 256 * i;            // 0..1023
            int row = idx >> 3;                 // 0..127 (M)
            int col = (idx & 7) << 2;           // 0..28  (K, float4)
            xr[i] = *(const float4*)(base + (size_t)row * IN_C + col);
        }
    };
    auto load_w = [&](int kt) {
        const float* base = wb + (size_t)kt * KT * FILTERS;
        #pragma unroll
        for (int i = 0; i < 4; ++i) {
            int idx  = tid + 256 * i;           // 0..1023
            int krow = idx >> 5;                // 0..31  (K)
            int col  = (idx & 31) << 2;         // 0..124 (N, float4)
            wr[i] = *(const float4*)(base + (size_t)krow * FILTERS + col);
        }
    };
    auto stage = [&]() {
        // x tile: split + store row-major [m][k]
        #pragma unroll
        for (int i = 0; i < 4; ++i) {
            int idx = tid + 256 * i;
            int row = idx >> 3;
            int col = (idx & 7) << 2;
            float vv[4] = {xr[i].x, xr[i].y, xr[i].z, xr[i].w};
            v4bf hv, lv;
            #pragma unroll
            for (int j = 0; j < 4; ++j) {
                __bf16 hh, ll;
                split_bf16(vv[j], hh, ll);
                hv[j] = hh;
                lv[j] = ll;
            }
            *(v4bf*)(&As_hi[row * LDK + col]) = hv;
            *(v4bf*)(&As_lo[row * LDK + col]) = lv;
        }
        // weight tile: split + store transposed [n][k]
        #pragma unroll
        for (int i = 0; i < 4; ++i) {
            int idx  = tid + 256 * i;
            int krow = idx >> 5;
            int col  = (idx & 31) << 2;
            float vv[4] = {wr[i].x, wr[i].y, wr[i].z, wr[i].w};
            #pragma unroll
            for (int j = 0; j < 4; ++j) {
                __bf16 hh, ll;
                split_bf16(vv[j], hh, ll);
                Wt_hi[(col + j) * LDK + krow] = hh;
                Wt_lo[(col + j) * LDK + krow] = ll;
            }
        }
    };

    load_x(0);
    load_w(0);

    for (int kt = 0; kt < NKSTEPS; ++kt) {
        __syncthreads();                 // LDS free from previous compute
        stage();
        if (kt + 1 < NKSTEPS) {          // prefetch next slice during compute
            load_x(kt + 1);
            load_w(kt + 1);
        }
        __syncthreads();                 // LDS tile visible

        // ---- fragment loads (ISA 7.12.2 layouts) ----
        AFrag a_hi[2], a_lo[2], b_hi[4], b_lo[4];
        #pragma unroll
        for (int mt = 0; mt < 2; ++mt) {
            int row = wm + mt * 16 + l16;
            const __bf16* ph = &As_hi[row * LDK];
            const __bf16* pl = &As_lo[row * LDK];
            a_hi[mt].h[0] = *(const v8bf*)(ph + 8 * h);        // K = 8h .. 8h+7
            a_hi[mt].h[1] = *(const v8bf*)(ph + 16 + 8 * h);   // K = 16+8h .. 23+8h
            a_lo[mt].h[0] = *(const v8bf*)(pl + 8 * h);
            a_lo[mt].h[1] = *(const v8bf*)(pl + 16 + 8 * h);
        }
        #pragma unroll
        for (int nt = 0; nt < 4; ++nt) {
            int rown = wn + nt * 16 + l16;
            const __bf16* qh = &Wt_hi[rown * LDK];    // K = 8h..8h+7, 16+8h..23+8h
            const __bf16* ql = &Wt_lo[rown * LDK];
            b_hi[nt].h[0] = *(const v8bf*)(qh + 8 * h);
            b_hi[nt].h[1] = *(const v8bf*)(qh + 16 + 8 * h);
            b_lo[nt].h[0] = *(const v8bf*)(ql + 8 * h);
            b_lo[nt].h[1] = *(const v8bf*)(ql + 16 + 8 * h);
        }

        // ---- bf16x3 WMMA: lo*hi + hi*lo + hi*hi, f32 accumulate ----
        #pragma unroll
        for (int mt = 0; mt < 2; ++mt) {
            #pragma unroll
            for (int nt = 0; nt < 4; ++nt) {
                v8f a = acc[mt][nt];
                a = __builtin_amdgcn_wmma_f32_16x16x32_bf16(
                        false, a_lo[mt].v, false, b_hi[nt].v, (short)0, a, false, false);
                a = __builtin_amdgcn_wmma_f32_16x16x32_bf16(
                        false, a_hi[mt].v, false, b_lo[nt].v, (short)0, a, false, false);
                a = __builtin_amdgcn_wmma_f32_16x16x32_bf16(
                        false, a_hi[mt].v, false, b_hi[nt].v, (short)0, a, false, false);
                acc[mt][nt] = a;
            }
        }
    }

    // ---- epilogue: bias add + store (C/D layout: lane n = l16, VGPR r -> M = r+8h) ----
    #pragma unroll
    for (int nt = 0; nt < 4; ++nt) {
        int n = bn + wn + nt * 16 + l16;
        float bv = bias[(size_t)c * FILTERS + n];
        #pragma unroll
        for (int mt = 0; mt < 2; ++mt) {
            #pragma unroll
            for (int r = 0; r < 8; ++r) {
                int m = bm + wm + mt * 16 + r + 8 * h;
                out[((size_t)b * PIX + m) * FILTERS + n] = acc[mt][nt][r] + bv;
            }
        }
    }
}

extern "C" void kernel_launch(void* const* d_in, const int* in_sizes, int n_in,
                              void* d_out, int out_size, void* d_ws, size_t ws_size,
                              hipStream_t stream) {
    const float* x    = (const float*)d_in[0];
    const int*   cls  = (const int*)d_in[1];
    const float* ktab = (const float*)d_in[2];
    const float* bias = (const float*)d_in[3];
    float* out = (float*)d_out;

    const int B = in_sizes[1];  // number of samples (cls element count)
    dim3 grid(FILTERS / NT, PIX / MT, B);
    dim3 block(256);
    hipLaunchKernelGGL(cond_conv11_wmma_bf16x3, grid, block, 0, stream,
                       x, cls, ktab, bias, out);
}
